// Transformer_58076547777168
// MI455X (gfx1250) — matrix operand order repaired and checked by reading, not compiled, not run on hardware
//
#include <hip/hip_runtime.h>

// ---------------------------------------------------------------------------
// CDNA5 (gfx1250) transformer forward.  bf16 WMMA GEMMs (async-LDS B staging)
// + flash attention.
// ---------------------------------------------------------------------------

typedef __attribute__((ext_vector_type(16))) __bf16 v16bf;
typedef __attribute__((ext_vector_type(8)))  float  v8f;

#define WMMA_BF16(A, B, C) \
    __builtin_amdgcn_wmma_f32_16x16x32_bf16(false, (A), false, (B), (short)0, (C), false, false)

__device__ __forceinline__ unsigned short f2bf(float f) {
    unsigned u = __float_as_uint(f);
    u += 0x7FFFu + ((u >> 16) & 1u);           // round-to-nearest-even
    return (unsigned short)(u >> 16);
}

union V16U {
    uint4          q[2];
    unsigned short s[16];
    v16bf          v;
};

// ---- async Global->LDS staging (guarded: falls back to sync copy) ---------
__device__ __forceinline__ void stage_b128(const unsigned short* g, unsigned short* l) {
#if __has_builtin(__builtin_amdgcn_global_load_async_to_lds_b128)
    typedef __attribute__((ext_vector_type(4))) int v4i;
    typedef __attribute__((address_space(1))) v4i* gp_t;   // global v4i*
    typedef __attribute__((address_space(3))) v4i* lp_t;   // LDS v4i*
    __builtin_amdgcn_global_load_async_to_lds_b128((gp_t)(g), (lp_t)(l), 0, 0);
#else
    *(uint4*)l = *(const uint4*)g;
#endif
}

template <int N>
__device__ __forceinline__ void wait_async_le() {
#if __has_builtin(__builtin_amdgcn_s_wait_asynccnt)
    __builtin_amdgcn_s_wait_asynccnt(N);
#else
    if (N == 0) asm volatile("s_wait_asynccnt 0x0" ::: "memory");
    else        asm volatile("s_wait_asynccnt 0x1" ::: "memory");
#endif
}

// ---------------------------------------------------------------------------
// fp32 [K,N] -> bf16 [N,K] transpose (LDS tiled), batched over leading dim.
// ---------------------------------------------------------------------------
__global__ __launch_bounds__(256) void transpose_bf16_kernel(
    const float* __restrict__ w, unsigned short* __restrict__ wT, int K, int N)
{
    __shared__ float tile[32][33];
    const int l  = blockIdx.z;
    const size_t base  = (size_t)l * K * N;
    const size_t baseT = (size_t)l * N * K;
    const int kb = blockIdx.y * 32, nb = blockIdx.x * 32;
    const int tx = threadIdx.x, ty = threadIdx.y;   // block (32,8)
    for (int i = ty; i < 32; i += 8)
        tile[i][tx] = w[base + (size_t)(kb + i) * N + nb + tx];
    __syncthreads();
    for (int i = ty; i < 32; i += 8)
        wT[baseT + (size_t)(nb + i) * K + kb + tx] = f2bf(tile[tx][i]);
}

// ---------------------------------------------------------------------------
// token + positional embedding -> fp32 residual stream
// ---------------------------------------------------------------------------
__global__ __launch_bounds__(256) void embed_kernel(
    const int* __restrict__ tok, const float* __restrict__ te,
    const float* __restrict__ pe, float* __restrict__ x, int SEQ)
{
    const int t = blockIdx.x, tid = threadIdx.x;
    const int pos = t % SEQ;
    const int id  = tok[t];
    for (int d = tid; d < 512; d += 256)
        x[(size_t)t * 512 + d] = te[(size_t)id * 512 + d] + pe[(size_t)pos * 512 + d];
}

// ---------------------------------------------------------------------------
// LayerNorm (D=512 fixed): fp32 in -> bf16 out (feeds GEMMs)
// ---------------------------------------------------------------------------
__global__ __launch_bounds__(256) void layernorm_kernel(
    const float* __restrict__ x, const float* __restrict__ g,
    const float* __restrict__ bta, unsigned short* __restrict__ out)
{
    const int row = blockIdx.x, tid = threadIdx.x;
    __shared__ float r1[256], r2[256];
    const float* xp = x + (size_t)row * 512;
    const float v0 = xp[tid], v1 = xp[tid + 256];
    r1[tid] = v0 + v1;
    r2[tid] = v0 * v0 + v1 * v1;
    __syncthreads();
    for (int off = 128; off > 0; off >>= 1) {
        if (tid < off) { r1[tid] += r1[tid + off]; r2[tid] += r2[tid + off]; }
        __syncthreads();
    }
    const float mu   = r1[0] * (1.0f / 512.0f);
    const float var  = r2[0] * (1.0f / 512.0f) - mu * mu;
    const float rstd = rsqrtf(var + 1e-6f);
    unsigned short* op = out + (size_t)row * 512;
    op[tid]       = f2bf((v0 - mu) * rstd * g[tid] + bta[tid]);
    op[tid + 256] = f2bf((v1 - mu) * rstd * g[tid + 256] + bta[tid + 256]);
}

// ---------------------------------------------------------------------------
// WMMA GEMM: C[M,N] = A[M,K](bf16) * BT[N,K]^T(bf16) + bias
//   block = 256 threads (8 waves) -> 128(M) x 64(N) tile.
//   B tile (64 rows x 32 K-halves) staged in LDS once per block, double
//   buffered via async Global->LDS; all 8 waves broadcast-read it.
//   mode 0: f32 out (+ optional residual)     mode 1: bf16 row-major (relu opt)
//   mode 2: bf16 [B,H,S,DK] (q/k layout)      mode 3: bf16 [B,H,DK,S] (v^T)
// ---------------------------------------------------------------------------
#define BSTRIDE 48              // halves per LDS row (96B: 32B-aligned reads)
#define BUF_H   (64 * BSTRIDE)  // halves per buffer

__global__ __launch_bounds__(256) void gemm_bf16_kernel(
    const unsigned short* __restrict__ A, const unsigned short* __restrict__ BT,
    const float* __restrict__ bias, const float* residual,
    float* outF, unsigned short* outB,
    int M, int N, int K, int mode, int relu)
{
    __shared__ unsigned short bsh[2 * BUF_H];   // 12 KB
    const int tid  = threadIdx.x;
    const int lane = tid & 31, wave = tid >> 5;
    const int hi = lane >> 4, ln = lane & 15;
    const int m0 = blockIdx.y * 128 + wave * 16;
    const int n0 = blockIdx.x * 64;

    const unsigned short* arow = A + (size_t)(m0 + ln) * K;

    // staging map: thread -> (row 0..63, 16B chunk 0..3) of the B tile
    const int srow = tid >> 2, schk = tid & 3;
    const unsigned short* gsrc = BT + (size_t)(n0 + srow) * K + schk * 8;
    unsigned short*       ldst = bsh + srow * BSTRIDE + schk * 8;

    stage_b128(gsrc, ldst);                       // fill buffer 0 (k0 = 0)

    v8f acc[4] = {};
    for (int k0 = 0; k0 < K; k0 += 32) {
        const int cur = (k0 >> 5) & 1;
        if (k0 + 32 < K) {
            stage_b128(gsrc + k0 + 32, ldst + (cur ^ 1) * BUF_H);
            wait_async_le<1>();
        } else {
            wait_async_le<0>();
        }
        __syncthreads();                          // B tile visible to all waves

        if (k0 + 64 < K) __builtin_prefetch(arow + k0 + 64, 0, 0);

        V16U a;
        a.q[0] = *(const uint4*)(arow + k0 + 16 * hi);
        a.q[1] = *(const uint4*)(arow + k0 + 16 * hi + 8);
        const unsigned short* bb = bsh + cur * BUF_H + 16 * hi;
#pragma unroll
        for (int j = 0; j < 4; j++) {
            v16bf b = *(const v16bf*)(bb + (16 * j + ln) * BSTRIDE);
            acc[j] = WMMA_BF16(a.v, b, acc[j]);
        }
        __syncthreads();                          // done reading this buffer
    }

#pragma unroll
    for (int j = 0; j < 4; j++) {
        const int n = n0 + 16 * j + ln;
        const float bv = bias ? bias[n] : 0.0f;
#pragma unroll
        for (int r = 0; r < 8; r++) {
            const int m = m0 + r + 8 * hi;
            float v = acc[j][r] + bv;
            if (relu) v = fmaxf(v, 0.0f);
            if (mode == 0) {
                const size_t idx = (size_t)m * N + n;
                float res = residual ? residual[idx] : 0.0f;
                outF[idx] = res + v;
            } else if (mode == 1) {
                outB[(size_t)m * N + n] = f2bf(v);
            } else {
                const int b = m >> 10, s = m & 1023;      // SEQ = 1024
                const int h = n >> 6,  dk = n & 63;       // DK  = 64
                const size_t idx = (mode == 2)
                    ? ((((size_t)b * 8 + h) * 1024 + s) * 64 + dk)
                    : ((((size_t)b * 8 + h) * 64 + dk) * 1024 + s);
                outB[idx] = f2bf(v);
            }
        }
    }
}

// ---------------------------------------------------------------------------
// Flash attention (WMMA): Q[BH,Sq,64] K[BH,Sk,64] V^T[BH,64,Sk] (all bf16)
// -> Out bf16 [B,Sq,512] row-major (ready as next GEMM's A operand).
// block = 128 threads (4 waves), wave owns 16 query rows; 32-key chunks.
// ---------------------------------------------------------------------------
__global__ __launch_bounds__(128) void attn_kernel(
    const unsigned short* __restrict__ Q, const unsigned short* __restrict__ Km,
    const unsigned short* __restrict__ VT, unsigned short* __restrict__ Out,
    int Sq, int Sk, int causal)
{
    __shared__ float lds[4 * 16 * 32];
    const int lane = threadIdx.x & 31, wave = threadIdx.x >> 5;
    const int hi = lane >> 4, ln = lane & 15;
    const int bh = blockIdx.y;
    const int b = bh >> 3, h = bh & 7;
    const int q0 = (blockIdx.x * 4 + wave) * 16;
    float* wlds = lds + wave * 512;

    const unsigned short* Qp = Q  + ((size_t)bh * Sq + q0) * 64;
    const unsigned short* Kp = Km + (size_t)bh * Sk * 64;
    const unsigned short* Vp = VT + (size_t)bh * 64 * Sk;

    // Q A-fragments (feat 0..31 and 32..63), loaded once
    const unsigned short* qrowp = Qp + ln * 64;
    V16U aq0, aq1;
    aq0.q[0] = *(const uint4*)(qrowp + 16 * hi);
    aq0.q[1] = *(const uint4*)(qrowp + 8 + 16 * hi);
    aq1.q[0] = *(const uint4*)(qrowp + 32 + 16 * hi);
    aq1.q[1] = *(const uint4*)(qrowp + 40 + 16 * hi);

    v8f o0 = {}, o1 = {}, o2 = {}, o3 = {};
    float mrow[8], lrow[8];
#pragma unroll
    for (int r = 0; r < 8; r++) { mrow[r] = -3.0e38f; lrow[r] = 0.0f; }

    const int limit = causal ? (q0 + 16) : Sk;
    for (int kc = 0; kc < limit; kc += 32) {
        // S = Q * K^T   (two 16x16 n-tiles, K-dim = 64 features = 2 WMMAs each)
        const unsigned short* kb = Kp + 16 * hi;
        if (kc + 32 < limit) {
            __builtin_prefetch(kb + (size_t)(kc + 32 + ln) * 64, 0, 0);
            __builtin_prefetch(Vp + (size_t)ln * Sk + kc + 32, 0, 0);
        }
        v16bf bk00 = *(const v16bf*)(kb + (size_t)(kc + ln) * 64);
        v16bf bk01 = *(const v16bf*)(kb + (size_t)(kc + ln) * 64 + 32);
        v16bf bk10 = *(const v16bf*)(kb + (size_t)(kc + 16 + ln) * 64);
        v16bf bk11 = *(const v16bf*)(kb + (size_t)(kc + 16 + ln) * 64 + 32);
        v8f s0 = {}, s1 = {};
        s0 = WMMA_BF16(aq0.v, bk00, s0);
        s0 = WMMA_BF16(aq1.v, bk01, s0);
        s1 = WMMA_BF16(aq0.v, bk10, s1);
        s1 = WMMA_BF16(aq1.v, bk11, s1);

        // online softmax over this 32-key chunk
        const float scale = 0.125f;   // 1/sqrt(64)
#pragma unroll
        for (int r = 0; r < 8; r++) {
            float v0 = s0[r] * scale, v1 = s1[r] * scale;
            if (causal) {
                const int qr = q0 + r + 8 * hi;
                if (kc + ln > qr)      v0 = -3.0e38f;
                if (kc + 16 + ln > qr) v1 = -3.0e38f;
            }
            float mx = fmaxf(v0, v1);
            mx = fmaxf(mx, __shfl_xor(mx, 1, 32));
            mx = fmaxf(mx, __shfl_xor(mx, 2, 32));
            mx = fmaxf(mx, __shfl_xor(mx, 4, 32));
            mx = fmaxf(mx, __shfl_xor(mx, 8, 32));
            const float newm = fmaxf(mrow[r], mx);
            const float c  = __expf(mrow[r] - newm);
            const float p0 = __expf(v0 - newm);
            const float p1 = __expf(v1 - newm);
            float rs = p0 + p1;
            rs += __shfl_xor(rs, 1, 32);
            rs += __shfl_xor(rs, 2, 32);
            rs += __shfl_xor(rs, 4, 32);
            rs += __shfl_xor(rs, 8, 32);
            lrow[r] = lrow[r] * c + rs;
            mrow[r] = newm;
            o0[r] *= c; o1[r] *= c; o2[r] *= c; o3[r] *= c;
            wlds[(r + 8 * hi) * 32 + ln]      = p0;   // C-layout -> LDS row-major
            wlds[(r + 8 * hi) * 32 + 16 + ln] = p1;
        }
        // wave-synchronous LDS visibility (no block barrier: waves diverge)
        asm volatile("s_wait_dscnt 0" ::: "memory");

        // re-read P in A-fragment layout and convert to bf16
        V16U ap;
#pragma unroll
        for (int hh = 0; hh < 16; hh++) {
            const int col = hh + 16 * hi;
            ap.s[hh] = f2bf(wlds[ln * 32 + col]);
        }

        // O += P * V   (V^T rows are features; 4 feature n-tiles)
        const unsigned short* vb = Vp + kc + 16 * hi;
        v16bf bv0 = *(const v16bf*)(vb + (size_t)(ln)      * Sk);
        v16bf bv1 = *(const v16bf*)(vb + (size_t)(16 + ln) * Sk);
        v16bf bv2 = *(const v16bf*)(vb + (size_t)(32 + ln) * Sk);
        v16bf bv3 = *(const v16bf*)(vb + (size_t)(48 + ln) * Sk);
        o0 = WMMA_BF16(ap.v, bv0, o0);
        o1 = WMMA_BF16(ap.v, bv1, o1);
        o2 = WMMA_BF16(ap.v, bv2, o2);
        o3 = WMMA_BF16(ap.v, bv3, o3);
    }

    // epilogue: normalize and scatter into [B,Sq,512] bf16
    unsigned short* op = Out + (size_t)b * Sq * 512 + (size_t)h * 64;
    v8f oarr[4] = {o0, o1, o2, o3};
#pragma unroll
    for (int j = 0; j < 4; j++) {
        const int n = j * 16 + ln;
#pragma unroll
        for (int r = 0; r < 8; r++) {
            const int qr = q0 + r + 8 * hi;
            op[(size_t)qr * 512 + n] = f2bf(oarr[j][r] / lrow[r]);
        }
    }
}

// ---------------------------------------------------------------------------
// In-place row-wise log-softmax over V
// ---------------------------------------------------------------------------
__global__ __launch_bounds__(256) void logsoftmax_kernel(float* __restrict__ p, int Vn)
{
    const int row = blockIdx.x, tid = threadIdx.x;
    float* rp = p + (size_t)row * Vn;
    __shared__ float red[256];
    float mx = -3.0e38f;
    for (int i = tid; i < Vn; i += 256) mx = fmaxf(mx, rp[i]);
    red[tid] = mx; __syncthreads();
    for (int off = 128; off > 0; off >>= 1) {
        if (tid < off) red[tid] = fmaxf(red[tid], red[tid + off]);
        __syncthreads();
    }
    mx = red[0]; __syncthreads();
    float s = 0.0f;
    for (int i = tid; i < Vn; i += 256) s += __expf(rp[i] - mx);
    red[tid] = s; __syncthreads();
    for (int off = 128; off > 0; off >>= 1) {
        if (tid < off) red[tid] += red[tid + off];
        __syncthreads();
    }
    const float lse = mx + __logf(red[0]);
    for (int i = tid; i < Vn; i += 256) rp[i] -= lse;
}

// ---------------------------------------------------------------------------
// Host orchestration
// ---------------------------------------------------------------------------
enum {
    I_SRC_TOK = 0, I_TGT_TOK, I_SRC_MASK, I_TGT_MASK, I_TOK_EMB, I_POS_EMB,
    E_WQ, E_BQ, E_WK, E_BK, E_WV, E_BV, E_WO, E_BO, E_W1, E_B1, E_W2, E_B2,
    E_LN1G, E_LN1B, E_LN2G, E_LN2B,
    D_WQ, D_BQ, D_WK, D_BK, D_WV, D_BV, D_WO, D_BO,
    D_CWQ, D_CBQ, D_CWK, D_CBK, D_CWV, D_CBV, D_CWO, D_CBO,
    D_W1, D_B1, D_W2, D_B2,
    D_LN1G, D_LN1B, D_LN2G, D_LN2B, D_LN3G, D_LN3B,
    ENC_LNG, ENC_LNB, DEC_LNG, DEC_LNB, GEN_W, GEN_B
};

extern "C" void kernel_launch(void* const* d_in, const int* in_sizes, int n_in,
                              void* d_out, int out_size, void* d_ws, size_t ws_size,
                              hipStream_t stream)
{
    (void)in_sizes; (void)n_in; (void)out_size; (void)ws_size;
    constexpr int Bn = 4, S = 1024, T = 1024, Dm = 512, Hh = 8, Ll = 4,
                  FF = 2048, Vv = 32000;
    constexpr int M = Bn * S;        // 4096 tokens (same for src and tgt)

    auto F = [&](int i) { return (const float*)d_in[i]; };

    // ---- workspace bump allocator ----
    char* base = (char*)d_ws; size_t off = 0;
    auto alloc  = [&](size_t bytes) { void* p = base + off; off = (off + bytes + 255) & ~(size_t)255; return p; };
    auto allocU = [&](size_t n) { return (unsigned short*)alloc(n * 2); };
    auto allocF = [&](size_t n) { return (float*)alloc(n * 4); };

    const size_t DD = (size_t)Dm * Dm, FD = (size_t)FF * Dm;
    unsigned short* eWqT = allocU(Ll * DD); unsigned short* eWkT = allocU(Ll * DD);
    unsigned short* eWvT = allocU(Ll * DD); unsigned short* eWoT = allocU(Ll * DD);
    unsigned short* eW1T = allocU(Ll * FD); unsigned short* eW2T = allocU(Ll * FD);
    unsigned short* dWqT = allocU(Ll * DD); unsigned short* dWkT = allocU(Ll * DD);
    unsigned short* dWvT = allocU(Ll * DD); unsigned short* dWoT = allocU(Ll * DD);
    unsigned short* cWqT = allocU(Ll * DD); unsigned short* cWkT = allocU(Ll * DD);
    unsigned short* cWvT = allocU(Ll * DD); unsigned short* cWoT = allocU(Ll * DD);
    unsigned short* dW1T = allocU(Ll * FD); unsigned short* dW2T = allocU(Ll * FD);
    unsigned short* genT = allocU((size_t)Vv * Dm);

    float*          x_f32   = allocF((size_t)M * Dm);
    float*          y_f32   = allocF((size_t)M * Dm);
    unsigned short* xn_bf   = allocU((size_t)M * Dm);
    unsigned short* mem_bf  = allocU((size_t)M * Dm);
    unsigned short* q_bf    = allocU((size_t)M * Dm);
    unsigned short* k_bf    = allocU((size_t)M * Dm);
    unsigned short* v_bf    = allocU((size_t)M * Dm);
    unsigned short* attn_bf = allocU((size_t)M * Dm);
    unsigned short* h_bf    = allocU((size_t)M * FF);

    auto xpose = [&](const float* w, unsigned short* wT, int K, int N, int batch) {
        transpose_bf16_kernel<<<dim3(N / 32, K / 32, batch), dim3(32, 8, 1), 0, stream>>>(w, wT, K, N);
    };
    auto gemm = [&](const unsigned short* A, const unsigned short* BT, const float* bias,
                    const float* res, float* oF, unsigned short* oB,
                    int Mq, int Nq, int Kq, int mode, int relu) {
        gemm_bf16_kernel<<<dim3(Nq / 64, Mq / 128), 256, 0, stream>>>(
            A, BT, bias, res, oF, oB, Mq, Nq, Kq, mode, relu);
    };
    auto lnorm = [&](const float* x, const float* g, const float* b, unsigned short* o) {
        layernorm_kernel<<<M, 256, 0, stream>>>(x, g, b, o);
    };
    auto attn = [&](const unsigned short* q, const unsigned short* k, const unsigned short* v,
                    unsigned short* o, int Sq, int Sk, int causal) {
        attn_kernel<<<dim3(Sq / 64, Bn * Hh), 128, 0, stream>>>(q, k, v, o, Sq, Sk, causal);
    };

    // ---- weight conversion / transpose (bf16) ----
    xpose(F(E_WQ),  eWqT, Dm, Dm, Ll); xpose(F(E_WK),  eWkT, Dm, Dm, Ll);
    xpose(F(E_WV),  eWvT, Dm, Dm, Ll); xpose(F(E_WO),  eWoT, Dm, Dm, Ll);
    xpose(F(E_W1),  eW1T, Dm, FF, Ll); xpose(F(E_W2),  eW2T, FF, Dm, Ll);
    xpose(F(D_WQ),  dWqT, Dm, Dm, Ll); xpose(F(D_WK),  dWkT, Dm, Dm, Ll);
    xpose(F(D_WV),  dWvT, Dm, Dm, Ll); xpose(F(D_WO),  dWoT, Dm, Dm, Ll);
    xpose(F(D_CWQ), cWqT, Dm, Dm, Ll); xpose(F(D_CWK), cWkT, Dm, Dm, Ll);
    xpose(F(D_CWV), cWvT, Dm, Dm, Ll); xpose(F(D_CWO), cWoT, Dm, Dm, Ll);
    xpose(F(D_W1),  dW1T, Dm, FF, Ll); xpose(F(D_W2),  dW2T, FF, Dm, Ll);
    xpose(F(GEN_W), genT, Dm, Vv, 1);

    // ---- encoder ----
    embed_kernel<<<M, 256, 0, stream>>>((const int*)d_in[I_SRC_TOK], F(I_TOK_EMB), F(I_POS_EMB), x_f32, S);
    for (int i = 0; i < Ll; i++) {
        const size_t dd = (size_t)i * DD, fd = (size_t)i * FD;
        lnorm(x_f32, F(E_LN1G) + i * Dm, F(E_LN1B) + i * Dm, xn_bf);
        gemm(xn_bf, eWqT + dd, F(E_BQ) + i * Dm, nullptr, nullptr, q_bf, M, Dm, Dm, 2, 0);
        gemm(xn_bf, eWkT + dd, F(E_BK) + i * Dm, nullptr, nullptr, k_bf, M, Dm, Dm, 2, 0);
        gemm(xn_bf, eWvT + dd, F(E_BV) + i * Dm, nullptr, nullptr, v_bf, M, Dm, Dm, 3, 0);
        attn(q_bf, k_bf, v_bf, attn_bf, S, S, 0);
        gemm(attn_bf, eWoT + dd, F(E_BO) + i * Dm, x_f32, x_f32, nullptr, M, Dm, Dm, 0, 0);
        lnorm(x_f32, F(E_LN2G) + i * Dm, F(E_LN2B) + i * Dm, xn_bf);
        gemm(xn_bf, eW1T + fd, F(E_B1) + i * FF, nullptr, nullptr, h_bf, M, FF, Dm, 1, 1);
        gemm(h_bf, eW2T + fd, F(E_B2) + i * Dm, x_f32, x_f32, nullptr, M, Dm, FF, 0, 0);
    }
    lnorm(x_f32, F(ENC_LNG), F(ENC_LNB), mem_bf);

    // ---- decoder ----
    embed_kernel<<<M, 256, 0, stream>>>((const int*)d_in[I_TGT_TOK], F(I_TOK_EMB), F(I_POS_EMB), y_f32, T);
    for (int i = 0; i < Ll; i++) {
        const size_t dd = (size_t)i * DD, fd = (size_t)i * FD;
        // masked self-attention
        lnorm(y_f32, F(D_LN1G) + i * Dm, F(D_LN1B) + i * Dm, xn_bf);
        gemm(xn_bf, dWqT + dd, F(D_BQ) + i * Dm, nullptr, nullptr, q_bf, M, Dm, Dm, 2, 0);
        gemm(xn_bf, dWkT + dd, F(D_BK) + i * Dm, nullptr, nullptr, k_bf, M, Dm, Dm, 2, 0);
        gemm(xn_bf, dWvT + dd, F(D_BV) + i * Dm, nullptr, nullptr, v_bf, M, Dm, Dm, 3, 0);
        attn(q_bf, k_bf, v_bf, attn_bf, T, T, 1);
        gemm(attn_bf, dWoT + dd, F(D_BO) + i * Dm, y_f32, y_f32, nullptr, M, Dm, Dm, 0, 0);
        // cross-attention against encoder memory
        lnorm(y_f32, F(D_LN2G) + i * Dm, F(D_LN2B) + i * Dm, xn_bf);
        gemm(xn_bf,  cWqT + dd, F(D_CBQ) + i * Dm, nullptr, nullptr, q_bf, M, Dm, Dm, 2, 0);
        gemm(mem_bf, cWkT + dd, F(D_CBK) + i * Dm, nullptr, nullptr, k_bf, M, Dm, Dm, 2, 0);
        gemm(mem_bf, cWvT + dd, F(D_CBV) + i * Dm, nullptr, nullptr, v_bf, M, Dm, Dm, 3, 0);
        attn(q_bf, k_bf, v_bf, attn_bf, T, S, 0);
        gemm(attn_bf, cWoT + dd, F(D_CBO) + i * Dm, y_f32, y_f32, nullptr, M, Dm, Dm, 0, 0);
        // FFN
        lnorm(y_f32, F(D_LN3G) + i * Dm, F(D_LN3B) + i * Dm, xn_bf);
        gemm(xn_bf, dW1T + fd, F(D_B1) + i * FF, nullptr, nullptr, h_bf, M, FF, Dm, 1, 1);
        gemm(h_bf, dW2T + fd, F(D_B2) + i * Dm, y_f32, y_f32, nullptr, M, Dm, FF, 0, 0);
    }

    // ---- generator + log-softmax ----
    lnorm(y_f32, F(DEC_LNG), F(DEC_LNB), xn_bf);
    gemm(xn_bf, genT, F(GEN_B), nullptr, (float*)d_out, nullptr, M, Vv, Dm, 0, 0);
    logsoftmax_kernel<<<M, 256, 0, stream>>>((float*)d_out, Vv);
}
